// Model_39676907881452
// MI455X (gfx1250) — matrix operand order repaired and checked by reading, not compiled, not run on hardware
//
#include <hip/hip_runtime.h>
#include <hip/hip_bf16.h>
#include <math.h>

// Problem constants (match reference)
#define B_  8
#define S_  2048
#define D_  512
#define DV_ 512

// Tiling
#define BM  64     // query rows per workgroup
#define BN  64     // key/value columns per iteration
#define NT  512    // threads per workgroup (16 wave32s)

// LDS row strides (bf16 elements) — padded to avoid 64-bank conflicts,
// kept multiples of 8 elems (16B) for b128 fragment loads.
#define LDQ 520
#define LDK 520
#define LDV 72
#define LDP 72

typedef __attribute__((ext_vector_type(16))) __bf16 v16bf;
typedef __attribute__((ext_vector_type(4)))  __bf16 v4bf;
typedef __attribute__((ext_vector_type(8)))  float  v8f;

__device__ __forceinline__ float rowmax16(float v) {
  v = fmaxf(v, __shfl_xor(v, 1));
  v = fmaxf(v, __shfl_xor(v, 2));
  v = fmaxf(v, __shfl_xor(v, 4));
  v = fmaxf(v, __shfl_xor(v, 8));
  return v;
}
__device__ __forceinline__ float rowsum16(float v) {
  v += __shfl_xor(v, 1);
  v += __shfl_xor(v, 2);
  v += __shfl_xor(v, 4);
  v += __shfl_xor(v, 8);
  return v;
}

// Convert a row-chunk of 64 f32 from global to bf16 in LDS (one thread).
__device__ __forceinline__ void cvt_row64(const float* __restrict__ src,
                                          __bf16* __restrict__ dst) {
#pragma unroll
  for (int j = 0; j < 16; ++j) {
    float4 f = ((const float4*)src)[j];
    v4bf h;
    h[0] = (__bf16)f.x; h[1] = (__bf16)f.y;
    h[2] = (__bf16)f.z; h[3] = (__bf16)f.w;
    *(v4bf*)(dst + j * 4) = h;
  }
}

// Load a 16x32 bf16 A fragment (ISA 16-bit A layout) from row-major LDS.
__device__ __forceinline__ v16bf load_afrag(const __bf16* base, int ldr,
                                            int ln, int lane, int ko) {
  const __bf16* ap = base + ln * ldr + ko + ((lane < 16) ? 0 : 8);
  v16bf a;
  *((uint4*)&a)     = *((const uint4*)ap);        // K = ko+{0..7} (+8 hi half)
  *((uint4*)&a + 1) = *((const uint4*)(ap + 16)); // K = ko+{16..23} (+8 hi half)
  return a;
}

// Load a 32x16 bf16 B fragment (ISA 16-bit B layout). LDS rows indexed by N,
// contiguous along K.
__device__ __forceinline__ v16bf load_bfrag(const __bf16* base, int ldr,
                                            int ln, int lane, int ko) {
  const __bf16* bp = base + ln * ldr + ko + ((lane < 16) ? 0 : 8);
  v16bf b;
  *((uint4*)&b)     = *((const uint4*)bp);        // K = ko+{0..7}   (+8 hi half)
  *((uint4*)&b + 1) = *((const uint4*)(bp + 16)); // K = ko+{16..23} (+8 hi half)
  return b;
}

__global__ __launch_bounds__(NT) void fa_fwd_kernel(
    const float* __restrict__ X,      // [B,S,D]
    const int*   __restrict__ mask,   // [S]
    const float* __restrict__ inv_scale, // [1]
    const float* __restrict__ Vmat,   // [DV,S]
    float* __restrict__ Out)          // [B,S,DV]
{
  extern __shared__ __align__(16) char smem[];
  __bf16* Qs    = (__bf16*)smem;            // 64  x LDQ
  __bf16* Ks    = Qs + BM * LDQ;            // 64  x LDK
  __bf16* Vs    = Ks + BN * LDK;            // 512 x LDV   (Vs[v][t])
  __bf16* Ps    = Vs + DV_ * LDV;           // 64  x LDP
  float*  biasL = (float*)(Ps + BM * LDP);  // 64
  float*  wmax  = biasL + BN;               // [4 rb][4 cw][16 rows]
  float*  wsum  = wmax + 4 * 4 * 16;        // [4 rb][4 cw][16 rows]

  const int tid   = threadIdx.x;
  const int lane  = tid & 31;
  const int w     = tid >> 5;
  const int rb    = w >> 2;                 // row-block 0..3 (16 rows each)
  const int cw    = w & 3;                  // col group (gemm1) / v-chunk (gemm2)
  const int half8 = (lane < 16) ? 0 : 8;    // C-layout: row = r + half8
  const int ln    = lane & 15;

  const int b  = blockIdx.y;
  const int s0 = blockIdx.x * BM;
  const float rscale = 1.0f / inv_scale[0];

  // ---- Load Q tile (64 x 512) as bf16 ----
  {
    const int row = tid >> 3;
    const int d0  = (tid & 7) * 64;
    cvt_row64(X + ((size_t)b * S_ + s0 + row) * D_ + d0, Qs + row * LDQ + d0);
  }

  float m_i[8], l_i[8];
  v8f Oacc[8];
#pragma unroll
  for (int r = 0; r < 8; ++r) { m_i[r] = -INFINITY; l_i[r] = 0.0f; }
#pragma unroll
  for (int vt = 0; vt < 8; ++vt)
    Oacc[vt] = (v8f){0.f, 0.f, 0.f, 0.f, 0.f, 0.f, 0.f, 0.f};

  for (int tb = 0; tb < S_; tb += BN) {
    __syncthreads();  // previous-iteration consumers done before overwriting LDS

    // ---- Stage K tile (64 rows of X[b]) and V^T tile (Vs[v][t]) ----
    {
      const int row = tid >> 3;
      const int d0  = (tid & 7) * 64;
      cvt_row64(X + ((size_t)b * S_ + tb + row) * D_ + d0, Ks + row * LDK + d0);
    }
    cvt_row64(Vmat + (size_t)tid * S_ + tb, Vs + tid * LDV);
    if (tid < BN) biasL[tid] = (mask[tb + tid] == 0) ? -1e30f : 0.0f;
    __syncthreads();

    // ---- GEMM1: this wave's 16x16 score tile over D=512 ----
    v8f sacc = (v8f){0.f, 0.f, 0.f, 0.f, 0.f, 0.f, 0.f, 0.f};
#pragma unroll
    for (int ks = 0; ks < D_ / 32; ++ks) {
      const int ko = ks * 32;
      v16bf a  = load_afrag(Qs + (rb * 16) * LDQ, LDQ, ln, lane, ko);
      v16bf bk = load_bfrag(Ks + (cw * 16) * LDK, LDK, ln, lane, ko);
      sacc = __builtin_amdgcn_wmma_f32_16x16x32_bf16(
          false, a, false, bk, (short)0, sacc, false, false);
    }

    // ---- scale + mask bias, per-wave row max ----
    const float bcol = biasL[cw * 16 + ln];
    float rmax[8];
#pragma unroll
    for (int r = 0; r < 8; ++r) {
      float s = sacc[r] * rscale + bcol;
      sacc[r] = s;
      rmax[r] = rowmax16(s);
    }
    if (lane == 0 || lane == 16) {
#pragma unroll
      for (int r = 0; r < 8; ++r)
        wmax[(rb * 4 + cw) * 16 + r + half8] = rmax[r];
    }
    __syncthreads();

    // ---- combine maxima across the 4 column-waves; online rescale factors ----
    float newm[8], alpha[8];
#pragma unroll
    for (int r = 0; r < 8; ++r) {
      const int row = r + half8;
      float bm = wmax[(rb * 4 + 0) * 16 + row];
      bm = fmaxf(bm, wmax[(rb * 4 + 1) * 16 + row]);
      bm = fmaxf(bm, wmax[(rb * 4 + 2) * 16 + row]);
      bm = fmaxf(bm, wmax[(rb * 4 + 3) * 16 + row]);
      const float nm = fmaxf(m_i[r], bm);
      alpha[r] = __expf(m_i[r] - nm);
      newm[r]  = nm;
    }

    // ---- P = exp(s - m); write bf16 P tile; per-wave row sums ----
#pragma unroll
    for (int r = 0; r < 8; ++r) {
      const float p = __expf(sacc[r] - newm[r]);
      Ps[(rb * 16 + r + half8) * LDP + cw * 16 + ln] = (__bf16)p;
      const float rs = rowsum16(p);
      if (lane == 0 || lane == 16)
        wsum[(rb * 4 + cw) * 16 + r + half8] = rs;
      m_i[r] = newm[r];
    }
    __syncthreads();

#pragma unroll
    for (int r = 0; r < 8; ++r) {
      const int row = r + half8;
      const float ls = wsum[(rb * 4 + 0) * 16 + row] +
                       wsum[(rb * 4 + 1) * 16 + row] +
                       wsum[(rb * 4 + 2) * 16 + row] +
                       wsum[(rb * 4 + 3) * 16 + row];
      l_i[r] = l_i[r] * alpha[r] + ls;
    }

    // ---- GEMM2: O[rb rows, cw*128 .. +128) += P(16x64) * V^T(64x512) ----
    const v16bf aP0 = load_afrag(Ps + (rb * 16) * LDP, LDP, ln, lane, 0);
    const v16bf aP1 = load_afrag(Ps + (rb * 16) * LDP, LDP, ln, lane, 32);
#pragma unroll
    for (int vt = 0; vt < 8; ++vt) {
#pragma unroll
      for (int r = 0; r < 8; ++r) Oacc[vt][r] *= alpha[r];
      const int vbase = cw * 128 + vt * 16;
      v16bf bv0 = load_bfrag(Vs + vbase * LDV, LDV, ln, lane, 0);
      Oacc[vt] = __builtin_amdgcn_wmma_f32_16x16x32_bf16(
          false, aP0, false, bv0, (short)0, Oacc[vt], false, false);
      v16bf bv1 = load_bfrag(Vs + vbase * LDV, LDV, ln, lane, 32);
      Oacc[vt] = __builtin_amdgcn_wmma_f32_16x16x32_bf16(
          false, aP1, false, bv1, (short)0, Oacc[vt], false, false);
    }
  }

  // ---- normalize and store ----
  float* outp = Out + ((size_t)b * S_ + s0 + rb * 16) * DV_;
#pragma unroll
  for (int r = 0; r < 8; ++r) {
    const float inv_l = 1.0f / l_i[r];
    const size_t rowoff = (size_t)(r + half8) * DV_;
#pragma unroll
    for (int vt = 0; vt < 8; ++vt)
      outp[rowoff + cw * 128 + vt * 16 + ln] = Oacc[vt][r] * inv_l;
  }
}

// Second tuple output: the mask, appended after B*S*DV floats (bit-copied).
__global__ void mask_copy_kernel(const int* __restrict__ m, int* __restrict__ dst) {
  const int i = blockIdx.x * 256 + threadIdx.x;
  if (i < S_) dst[i] = m[i];
}

extern "C" void kernel_launch(void* const* d_in, const int* in_sizes, int n_in,
                              void* d_out, int out_size, void* d_ws, size_t ws_size,
                              hipStream_t stream) {
  (void)in_sizes; (void)n_in; (void)out_size; (void)d_ws; (void)ws_size;
  const float* X    = (const float*)d_in[0];
  const int*   mask = (const int*)d_in[1];
  const float* inv  = (const float*)d_in[2];
  const float* Vm   = (const float*)d_in[3];
  float* Out = (float*)d_out;

  const size_t lds_bytes =
      (size_t)(BM * LDQ + BN * LDK + DV_ * LDV + BM * LDP) * sizeof(__bf16) +
      (size_t)(BN + 4 * 4 * 16 + 4 * 4 * 16) * sizeof(float);  // ~212 KB < 320 KB

  dim3 grid(S_ / BM, B_);
  fa_fwd_kernel<<<grid, NT, lds_bytes, stream>>>(X, mask, inv, Vm, Out);

  int* mask_out = (int*)(Out + (size_t)B_ * S_ * DV_);
  mask_copy_kernel<<<(S_ + 255) / 256, 256, 0, stream>>>(mask, mask_out);
}
